// StabilityPredictor_13443247636502
// MI455X (gfx1250) — matrix operand order repaired and checked by reading, not compiled, not run on hardware
//
#include <hip/hip_runtime.h>
#include <hip/hip_bf16.h>
#include <math.h>

typedef __attribute__((ext_vector_type(16))) _Float16 v16h;
typedef __attribute__((ext_vector_type(8)))  float    v8f;

// global-address-space pointer types: force global_load_* (not flat_load_*) for the
// weight stream, so it is tracked by LOADcnt only (flat also ties up the LDS path / DScnt).
typedef __attribute__((address_space(1))) const _Float16* gh_p;
typedef __attribute__((address_space(1))) const v16h*     gv16h_p;

#define BATCH 4
#define LEN   1024
#define KNN   32
#define HDIM  128
#define FFD   512
#define EDGE_IN 32

// ---------------- helpers ----------------

static __device__ inline float gelu_exact(float x) {
    return 0.5f * x * (1.0f + erff(x * 0.70710678118654752440f));
}

static __device__ inline float blockSum128(float v, volatile float* red, int tid) {
    red[tid] = v;
    __syncthreads();
    for (int s = 64; s > 0; s >>= 1) {
        if (tid < s) red[tid] += red[tid + s];
        __syncthreads();
    }
    float r = red[0];
    __syncthreads();
    return r;
}

// ---- fragment-major addressing (CDNA5 16x16x32 f16 WMMA layouts) ----
//
// A-matrix 16x32 f16 register layout (per ISA):
//   lanes 0-15  (kg=0): M=lane,    elems 0..7 <-> K=0..7,   elems 8..15 <-> K=16..23
//   lanes 16-31 (kg=1): M=lane-16, elems 0..7 <-> K=8..15,  elems 8..15 <-> K=24..31
// A buffers (LDS) are stored fragment-major: frag f=(kt*2+mt), then lane, then elem,
// so a fragment load is one contiguous 32-byte v16h per lane.
static __device__ inline int afrag_off(int row, int col) {
    int mt = row >> 4, m = row & 15;
    int kt = col >> 5, kk = col & 31;
    int kg = (kk >> 3) & 1;
    int elem = (kk < 16) ? (kk - 8 * kg) : (kk - 8 - 8 * kg);
    int lane = m + 16 * kg;
    return ((kt * 2 + mt) * 32 + lane) * 16 + elem;
}

static __device__ inline v8f wmma_f16(v16h a, v16h b, v8f c) {
    return __builtin_amdgcn_wmma_f32_16x16x32_f16(
        /*neg_a=*/false, a, /*neg_b=*/false, b,
        /*c_mod=*/(short)0, c, /*reuse_a=*/false, /*reuse_b=*/false);
}

// GEMM: [32 x KT] (LDS f16, A-fragment-major) x [KT x 128] (global f16, B-fragment-major)
// 4 waves; each wave computes 4 of the 16 (2x8) 16x16 output tiles.
// outH written in A-fragment-major layout (feeds the next GEMM); outF row-major f32.
template <int KT, bool DO_GELU, bool TO_HALF>
static __device__ inline void gemm_32xKx128(const _Float16* __restrict__ Afrag,
                                            const _Float16* __restrict__ Wfrag,
                                            const float* __restrict__ bias,
                                            _Float16* outH, float* outF,
                                            int wave, int lane) {
    // A lives in LDS; help the backend use direct ds addressing.
    __builtin_assume(Afrag != nullptr);
    __builtin_assume(__builtin_amdgcn_is_shared((const void*)Afrag));

    // W lives in global memory: force address_space(1) so loads lower to global_load_b128.
    gh_p W = (gh_p)(unsigned long long)(const void*)Wfrag;

    constexpr int NKT = KT / 32;
    int mt     = wave >> 1;           // 0..1
    int ntBase = (wave & 1) * 4;      // 0 or 4
    v8f acc[4] = {};
#pragma unroll
    for (int kt = 0; kt < NKT; ++kt) {
        if (kt + 1 < NKT) {
            // pull next k-step's weight fragments toward the WGP (global prefetch)
            __builtin_prefetch(Wfrag + ((kt + 1) * 8 + ntBase) * 32 * 16, 0, 3);
        }
        v16h a = *(const v16h*)(Afrag + ((kt * 2 + mt) * 32 + lane) * 16);
#pragma unroll
        for (int t = 0; t < 4; ++t) {
            v16h b = *(gv16h_p)(W + ((kt * 8 + ntBase + t) * 32 + lane) * 16);
            acc[t] = wmma_f16(a, b, acc[t]);
        }
    }
    int n  = lane & 15;
    int kg = lane >> 4;
    if (TO_HALF) {
        __builtin_assume(outH != nullptr);
        __builtin_assume(__builtin_amdgcn_is_shared((const void*)outH));
    } else {
        __builtin_assume(outF != nullptr);
        __builtin_assume(__builtin_amdgcn_is_shared((const void*)outF));
    }
#pragma unroll
    for (int t = 0; t < 4; ++t) {
        int nc = (ntBase + t) * 16 + n;
        float bv = bias[nc];
#pragma unroll
        for (int r = 0; r < 8; ++r) {
            int m = mt * 16 + r + 8 * kg;
            float x = acc[t][r] + bv;
            if (DO_GELU) x = gelu_exact(x);
            if (TO_HALF) outH[afrag_off(m, nc)] = (_Float16)x;
            else         outF[m * 128 + nc] = x;
        }
    }
}

// ---------------- kernels ----------------

// Swizzle f32 weight [Kt x 128] (row-major) into f16 B-fragment-major layout:
// out[((kt*8+nt)*32 + lane)*16 + elem], lane = n%16 + 16*kg, k = kt*32 + (elem<8 ? 8*kg+elem : 16+8*kg+elem-8), col = nt*16 + n%16
__global__ void swizzle_w_kernel(const float* __restrict__ in, _Float16* __restrict__ out, int total) {
    int o = blockIdx.x * 256 + threadIdx.x;
    if (o >= total) return;
    int elem = o & 15;
    int lane = (o >> 4) & 31;
    int f    = o >> 9;
    int nt = f & 7;
    int kt = f >> 3;
    int n  = nt * 16 + (lane & 15);
    int kg = lane >> 4;
    int k  = kt * 32 + (elem < 8 ? 8 * kg + elem : 16 + 8 * kg + (elem - 8));
    out[o] = (_Float16)in[k * 128 + n];
}

// One block per (b,i): distances, top-K (lexicographic argmin), RBF+pos features, LN, h_E projection
__global__ void featurize_kernel(const float* __restrict__ X, const float* __restrict__ mask,
                                 const int* __restrict__ residx, const int* __restrict__ chain,
                                 const float* __restrict__ pos_w, const float* __restrict__ pos_b,
                                 const float* __restrict__ normE_s, const float* __restrict__ normE_b,
                                 const float* __restrict__ We_w, const float* __restrict__ We_b,
                                 int* __restrict__ Eidx, float* __restrict__ maskAtt,
                                 float* __restrict__ hE) {
    __shared__ float sD[LEN];
    __shared__ float sRv[256];
    __shared__ int   sRi[256];
    __shared__ int   sIdx[KNN];
    __shared__ float sDnn[KNN];
    __shared__ int   sPos[KNN];
    __shared__ float sFeat[KNN * EDGE_IN];

    int node = blockIdx.x;           // b*LEN + i
    int b = node >> 10;
    int tid = threadIdx.x;

    float xi0 = X[node * 3 + 0], xi1 = X[node * 3 + 1], xi2 = X[node * 3 + 2];
    float mi  = mask[node];

    for (int j = tid; j < LEN; j += 256) {
        int nj = b * LEN + j;
        float dx = xi0 - X[nj * 3 + 0];
        float dy = xi1 - X[nj * 3 + 1];
        float dz = xi2 - X[nj * 3 + 2];
        float d2 = dx * dx + dy * dy + dz * dz;
        sD[j] = sqrtf(d2 + 1e-6f) + (1.0f - mi * mask[nj]) * 1e4f;
    }
    __syncthreads();

    for (int k = 0; k < KNN; ++k) {
        float bv = 3.0e38f;
        int   bj = 0x7fffffff;
        for (int j = tid; j < LEN; j += 256) {
            float v = sD[j];
            if (v < bv || (v == bv && j < bj)) { bv = v; bj = j; }
        }
        sRv[tid] = bv; sRi[tid] = bj;
        __syncthreads();
        for (int s = 128; s > 0; s >>= 1) {
            if (tid < s) {
                float vo = sRv[tid + s]; int io = sRi[tid + s];
                if (vo < sRv[tid] || (vo == sRv[tid] && io < sRi[tid])) {
                    sRv[tid] = vo; sRi[tid] = io;
                }
            }
            __syncthreads();
        }
        if (tid == 0) {
            sIdx[k] = sRi[0];
            sDnn[k] = sRv[0];
            sD[sRi[0]] = 3.0e38f;
        }
        __syncthreads();
    }

    if (tid < KNN) {
        int j  = sIdx[tid];
        int nj = b * LEN + j;
        int off  = residx[node] - residx[nj];
        bool same = (chain[node] == chain[nj]);
        int offc = off < -32 ? -32 : (off > 32 ? 32 : off);
        sPos[tid] = same ? (offc + 32) : 65;
        maskAtt[node * KNN + tid] = mi * mask[nj];
        Eidx[node * KNN + tid] = j;
    }
    __syncthreads();

    for (int o = tid; o < KNN * EDGE_IN; o += 256) {
        int k = o >> 5, f = o & 31;
        float v;
        if (f < 16) {
            v = pos_w[sPos[k] * 16 + f] + pos_b[f];
        } else {
            int r = f - 16;
            float mu = 2.0f + (20.0f / 15.0f) * (float)r;
            float t = (sDnn[k] - mu) / 1.25f;
            v = expf(-t * t);
        }
        sFeat[k * EDGE_IN + f] = v;
    }
    __syncthreads();

    if (tid < KNN) {
        float s = 0.f;
        for (int f = 0; f < EDGE_IN; ++f) s += sFeat[tid * EDGE_IN + f];
        float m = s / (float)EDGE_IN;
        float q = 0.f;
        for (int f = 0; f < EDGE_IN; ++f) { float d = sFeat[tid * EDGE_IN + f] - m; q += d * d; }
        float rs = rsqrtf(q / (float)EDGE_IN + 1e-5f);
        for (int f = 0; f < EDGE_IN; ++f)
            sFeat[tid * EDGE_IN + f] = (sFeat[tid * EDGE_IN + f] - m) * rs * normE_s[f] + normE_b[f];
    }
    __syncthreads();

    for (int o = tid; o < KNN * HDIM; o += 256) {
        int k = o >> 7, h = o & 127;
        float acc = We_b[h];
        for (int f = 0; f < EDGE_IN; ++f) acc += sFeat[k * EDGE_IN + f] * We_w[f * HDIM + h];
        hE[(size_t)node * KNN * HDIM + o] = acc;
    }
}

// Node message + FF update. One block (128 thr = 4 waves) per node.
__global__ void message_kernel(const float* __restrict__ hVin, float* __restrict__ hVout,
                               const float* __restrict__ hE,
                               const int* __restrict__ Eidx, const float* __restrict__ maskAtt,
                               const float* __restrict__ mask,
                               const _Float16* __restrict__ mW1h, const float* __restrict__ mb1,
                               const _Float16* __restrict__ mW2h, const float* __restrict__ mb2,
                               const _Float16* __restrict__ mW3h, const float* __restrict__ mb3,
                               const float* __restrict__ n1_s, const float* __restrict__ n1_b,
                               const float* __restrict__ fWin, const float* __restrict__ fbin,
                               const float* __restrict__ fWout, const float* __restrict__ fbout,
                               const float* __restrict__ n2_s, const float* __restrict__ n2_b) {
    __shared__ __align__(32) _Float16 sEV[KNN * 3 * HDIM];   // A-fragment-major, 32x384
    __shared__ __align__(32) _Float16 sM[2][KNN * HDIM];     // A-fragment-major, 32x128
    __shared__ float    sOut[KNN * HDIM];                    // row-major f32
    __shared__ float    sHV[HDIM], sHV2[HDIM], sFF[FFD], sRed[HDIM];
    __shared__ int      sIdx[KNN];
    __shared__ float    sMa[KNN];

    int node = blockIdx.x;
    int tid = threadIdx.x;
    int wave = tid >> 5, lane = tid & 31;
    int bbase = (node >> 10) << 10;

    if (tid < KNN) {
        sIdx[tid] = Eidx[node * KNN + tid];
        sMa[tid]  = maskAtt[node * KNN + tid];
    }
    float hv_i = hVin[node * HDIM + tid];
    sHV[tid] = hv_i;
    __syncthreads();

    // build h_EV = [h_i | h_E | h_j] as f16, directly in A-fragment-major layout
    _Float16 hih = (_Float16)hv_i;
    for (int r = 0; r < KNN; ++r) sEV[afrag_off(r, tid)] = hih;
    const float* hEn = hE + (size_t)node * KNN * HDIM;
    for (int o = tid; o < KNN * HDIM; o += 128) {
        int r = o >> 7, c = o & 127;
        sEV[afrag_off(r, 128 + c)] = (_Float16)hEn[o];
    }
    for (int r = 0; r < KNN; ++r) {
        int j = sIdx[r];
        sEV[afrag_off(r, 256 + tid)] = (_Float16)hVin[(bbase + j) * HDIM + tid];
    }
    __syncthreads();

    gemm_32xKx128<384, true,  true >(sEV,   mW1h, mb1, sM[0], nullptr, wave, lane);
    __syncthreads();
    gemm_32xKx128<128, true,  true >(sM[0], mW2h, mb2, sM[1], nullptr, wave, lane);
    __syncthreads();
    gemm_32xKx128<128, false, false>(sM[1], mW3h, mb3, nullptr, sOut,  wave, lane);
    __syncthreads();

    // masked neighbor sum / 30
    float dh = 0.f;
    for (int r = 0; r < KNN; ++r) dh += sOut[r * HDIM + tid] * sMa[r];
    dh *= (1.0f / 30.0f);
    float x = sHV[tid] + dh;

    // LN1
    float mean = blockSum128(x, sRed, tid) / (float)HDIM;
    float d = x - mean;
    float var = blockSum128(d * d, sRed, tid) / (float)HDIM;
    float y = d * rsqrtf(var + 1e-5f) * n1_s[tid] + n1_b[tid];
    sHV2[tid] = y;
    __syncthreads();

    // FF: 128 -> 512 (gelu) -> 128
    for (int o = tid; o < FFD; o += 128) {
        float a = fbin[o];
        for (int c = 0; c < HDIM; ++c) a += sHV2[c] * fWin[c * FFD + o];
        sFF[o] = gelu_exact(a);
    }
    __syncthreads();
    float a2 = fbout[tid];
    for (int c = 0; c < FFD; ++c) a2 += sFF[c] * fWout[c * HDIM + tid];
    float x2 = y + a2;

    // LN2 + mask
    float mean2 = blockSum128(x2, sRed, tid) / (float)HDIM;
    float d2 = x2 - mean2;
    float var2 = blockSum128(d2 * d2, sRed, tid) / (float)HDIM;
    float y2 = d2 * rsqrtf(var2 + 1e-5f) * n2_s[tid] + n2_b[tid];
    hVout[node * HDIM + tid] = y2 * mask[node];
}

// Edge update: rebuild h_EV from updated h_V, 3-layer MLP, residual + per-edge LN into hE (in place).
__global__ void edge_kernel(const float* __restrict__ hV, float* __restrict__ hE,
                            const int* __restrict__ Eidx,
                            const _Float16* __restrict__ eW1h, const float* __restrict__ eb1,
                            const _Float16* __restrict__ eW2h, const float* __restrict__ eb2,
                            const _Float16* __restrict__ eW3h, const float* __restrict__ eb3,
                            const float* __restrict__ n3_s, const float* __restrict__ n3_b) {
    __shared__ __align__(32) _Float16 sEV[KNN * 3 * HDIM];
    __shared__ __align__(32) _Float16 sM[2][KNN * HDIM];
    __shared__ float    sOut[KNN * HDIM];
    __shared__ int      sIdx[KNN];
    __shared__ float    sPa[128], sPb[128];
    __shared__ float    sMean[KNN], sRstd[KNN];

    int node = blockIdx.x;
    int tid = threadIdx.x;
    int wave = tid >> 5, lane = tid & 31;
    int bbase = (node >> 10) << 10;

    if (tid < KNN) sIdx[tid] = Eidx[node * KNN + tid];
    float hv_i = hV[node * HDIM + tid];
    __syncthreads();

    _Float16 hih = (_Float16)hv_i;
    for (int r = 0; r < KNN; ++r) sEV[afrag_off(r, tid)] = hih;
    float* hEn = hE + (size_t)node * KNN * HDIM;
    for (int o = tid; o < KNN * HDIM; o += 128) {
        int r = o >> 7, c = o & 127;
        sEV[afrag_off(r, 128 + c)] = (_Float16)hEn[o];
    }
    for (int r = 0; r < KNN; ++r) {
        int j = sIdx[r];
        sEV[afrag_off(r, 256 + tid)] = (_Float16)hV[(bbase + j) * HDIM + tid];
    }
    __syncthreads();

    gemm_32xKx128<384, true,  true >(sEV,   eW1h, eb1, sM[0], nullptr, wave, lane);
    __syncthreads();
    gemm_32xKx128<128, true,  true >(sM[0], eW2h, eb2, sM[1], nullptr, wave, lane);
    __syncthreads();
    gemm_32xKx128<128, false, false>(sM[1], eW3h, eb3, nullptr, sOut,  wave, lane);
    __syncthreads();

    // residual + LN per edge row (4 threads per row)
    int rid = tid >> 2, sub = tid & 3;
    float ps = 0.f, pq = 0.f;
    for (int c = sub * 32; c < sub * 32 + 32; ++c) {
        float v = hEn[rid * HDIM + c] + sOut[rid * HDIM + c];
        sOut[rid * HDIM + c] = v;
        ps += v; pq += v * v;
    }
    sPa[tid] = ps; sPb[tid] = pq;
    __syncthreads();
    if (sub == 0) {
        float S = sPa[tid] + sPa[tid + 1] + sPa[tid + 2] + sPa[tid + 3];
        float Q = sPb[tid] + sPb[tid + 1] + sPb[tid + 2] + sPb[tid + 3];
        float m = S / (float)HDIM;
        sMean[rid] = m;
        sRstd[rid] = rsqrtf(Q / (float)HDIM - m * m + 1e-5f);
    }
    __syncthreads();
    float m = sMean[rid], rs = sRstd[rid];
    for (int c = sub * 32; c < sub * 32 + 32; ++c)
        hEn[rid * HDIM + c] = (sOut[rid * HDIM + c] - m) * rs * n3_s[c] + n3_b[c];
}

// Attention pooling + MLP head. One block per batch element.
__global__ void pool_head_kernel(const float* __restrict__ hV, const float* __restrict__ mask,
                                 const float* __restrict__ aW1, const float* __restrict__ ab1,
                                 const float* __restrict__ aW2, const float* __restrict__ ab2,
                                 const float* __restrict__ hW1, const float* __restrict__ hb1,
                                 const float* __restrict__ hW2, const float* __restrict__ hb2,
                                 float* __restrict__ out) {
    __shared__ float sAttn[LEN];
    __shared__ float sRed[256];
    __shared__ float sG[HDIM];
    __shared__ float sU[HDIM];

    int b = blockIdx.x;
    int tid = threadIdx.x;

    for (int l = tid; l < LEN; l += 256) {
        const float* hv = hV + (size_t)(b * LEN + l) * HDIM;
        float a = ab2[0];
        for (int j = 0; j < 64; ++j) {
            float u = ab1[j];
            for (int c = 0; c < HDIM; ++c) u += hv[c] * aW1[c * 64 + j];
            a += tanhf(u) * aW2[j];
        }
        if (mask[b * LEN + l] == 0.0f) a = -1e9f;
        sAttn[l] = a;
    }
    __syncthreads();

    float mx = -3.0e38f;
    for (int l = tid; l < LEN; l += 256) mx = fmaxf(mx, sAttn[l]);
    sRed[tid] = mx; __syncthreads();
    for (int s = 128; s > 0; s >>= 1) { if (tid < s) sRed[tid] = fmaxf(sRed[tid], sRed[tid + s]); __syncthreads(); }
    float M = sRed[0]; __syncthreads();

    float ssum = 0.f;
    for (int l = tid; l < LEN; l += 256) ssum += expf(sAttn[l] - M);
    sRed[tid] = ssum; __syncthreads();
    for (int s = 128; s > 0; s >>= 1) { if (tid < s) sRed[tid] += sRed[tid + s]; __syncthreads(); }
    float S = sRed[0]; __syncthreads();

    for (int l = tid; l < LEN; l += 256) sAttn[l] = expf(sAttn[l] - M) / S;
    __syncthreads();

    if (tid < HDIM) {
        float g = 0.f;
        for (int l = 0; l < LEN; ++l) g += sAttn[l] * hV[(size_t)(b * LEN + l) * HDIM + tid];
        sG[tid] = g;
    }
    __syncthreads();
    if (tid < HDIM) {
        float u = hb1[tid];
        for (int c = 0; c < HDIM; ++c) u += sG[c] * hW1[c * HDIM + tid];
        sU[tid] = gelu_exact(u);
    }
    __syncthreads();
    if (tid == 0) {
        float d = hb2[0];
        for (int c = 0; c < HDIM; ++c) d += sU[c] * hW2[c];
        out[b] = d;
    }
}

// ---------------- launch ----------------

extern "C" void kernel_launch(void* const* d_in, const int* in_sizes, int n_in,
                              void* d_out, int out_size, void* d_ws, size_t ws_size,
                              hipStream_t stream) {
    (void)in_sizes; (void)n_in; (void)out_size; (void)ws_size;

    const float* X       = (const float*)d_in[0];
    const float* mask    = (const float*)d_in[1];
    const int*   residx  = (const int*)d_in[4];
    const int*   chain   = (const int*)d_in[5];
    const float* pos_w   = (const float*)d_in[6];
    const float* pos_b   = (const float*)d_in[7];
    const float* normE_s = (const float*)d_in[8];
    const float* normE_b = (const float*)d_in[9];
    const float* We_w    = (const float*)d_in[10];
    const float* We_b    = (const float*)d_in[11];
    const float* mW1     = (const float*)d_in[12];
    const float* mb1     = (const float*)d_in[13];
    const float* mW2     = (const float*)d_in[14];
    const float* mb2     = (const float*)d_in[15];
    const float* mW3     = (const float*)d_in[16];
    const float* mb3     = (const float*)d_in[17];
    const float* n1_s    = (const float*)d_in[18];
    const float* n1_b    = (const float*)d_in[19];
    const float* fWin    = (const float*)d_in[20];
    const float* fbin    = (const float*)d_in[21];
    const float* fWout   = (const float*)d_in[22];
    const float* fbout   = (const float*)d_in[23];
    const float* n2_s    = (const float*)d_in[24];
    const float* n2_b    = (const float*)d_in[25];
    const float* eW1     = (const float*)d_in[26];
    const float* eb1     = (const float*)d_in[27];
    const float* eW2     = (const float*)d_in[28];
    const float* eb2     = (const float*)d_in[29];
    const float* eW3     = (const float*)d_in[30];
    const float* eb3     = (const float*)d_in[31];
    const float* n3_s    = (const float*)d_in[32];
    const float* n3_b    = (const float*)d_in[33];
    const float* aW1     = (const float*)d_in[34];
    const float* ab1     = (const float*)d_in[35];
    const float* aW2     = (const float*)d_in[36];
    const float* ab2     = (const float*)d_in[37];
    const float* hW1     = (const float*)d_in[38];
    const float* hb1     = (const float*)d_in[39];
    const float* hW2     = (const float*)d_in[40];
    const float* hb2     = (const float*)d_in[41];

    const int NODES = BATCH * LEN;                      // 4096
    char* ws = (char*)d_ws;
    size_t off = 0;
    int*        Eidx    = (int*)(ws + off);       off += (size_t)NODES * KNN * 4;          // 512 KB
    float*      maskAtt = (float*)(ws + off);     off += (size_t)NODES * KNN * 4;          // 512 KB
    float*      hE      = (float*)(ws + off);     off += (size_t)NODES * KNN * HDIM * 4;   // 64 MB
    float*      hVa     = (float*)(ws + off);     off += (size_t)NODES * HDIM * 4;         // 2 MB
    float*      hVb     = (float*)(ws + off);     off += (size_t)NODES * HDIM * 4;         // 2 MB
    _Float16*   mW1h    = (_Float16*)(ws + off);  off += (size_t)3 * 384 * 128 * 2;
    _Float16*   mW2h    = (_Float16*)(ws + off);  off += (size_t)3 * 128 * 128 * 2;
    _Float16*   mW3h    = (_Float16*)(ws + off);  off += (size_t)3 * 128 * 128 * 2;
    _Float16*   eW1h    = (_Float16*)(ws + off);  off += (size_t)3 * 384 * 128 * 2;
    _Float16*   eW2h    = (_Float16*)(ws + off);  off += (size_t)3 * 128 * 128 * 2;
    _Float16*   eW3h    = (_Float16*)(ws + off);  off += (size_t)3 * 128 * 128 * 2;

    // swizzle weights into f16 B-fragment-major layout (per layer)
    auto swz = [&](const float* src, _Float16* dst, int kt_rows, int layer) {
        int total = kt_rows * 128;
        swizzle_w_kernel<<<(total + 255) / 256, 256, 0, stream>>>(
            src + (size_t)layer * total, dst + (size_t)layer * total, total);
    };
    for (int l = 0; l < 3; ++l) {
        swz(mW1, mW1h, 384, l);
        swz(mW2, mW2h, 128, l);
        swz(mW3, mW3h, 128, l);
        swz(eW1, eW1h, 384, l);
        swz(eW2, eW2h, 128, l);
        swz(eW3, eW3h, 128, l);
    }

    hipMemsetAsync(hVa, 0, (size_t)NODES * HDIM * 4, stream);

    featurize_kernel<<<NODES, 256, 0, stream>>>(X, mask, residx, chain,
                                                pos_w, pos_b, normE_s, normE_b,
                                                We_w, We_b, Eidx, maskAtt, hE);

    float* hin  = hVa;
    float* hout = hVb;
    for (int l = 0; l < 3; ++l) {
        message_kernel<<<NODES, 128, 0, stream>>>(hin, hout, hE, Eidx, maskAtt, mask,
            mW1h + (size_t)l * 384 * 128, mb1 + l * 128,
            mW2h + (size_t)l * 128 * 128, mb2 + l * 128,
            mW3h + (size_t)l * 128 * 128, mb3 + l * 128,
            n1_s + l * 128, n1_b + l * 128,
            fWin + (size_t)l * 128 * 512, fbin + l * 512,
            fWout + (size_t)l * 512 * 128, fbout + l * 128,
            n2_s + l * 128, n2_b + l * 128);
        edge_kernel<<<NODES, 128, 0, stream>>>(hout, hE, Eidx,
            eW1h + (size_t)l * 384 * 128, eb1 + l * 128,
            eW2h + (size_t)l * 128 * 128, eb2 + l * 128,
            eW3h + (size_t)l * 128 * 128, eb3 + l * 128,
            n3_s + l * 128, n3_b + l * 128);
        float* t = hin; hin = hout; hout = t;
    }

    pool_head_kernel<<<BATCH, 256, 0, stream>>>(hin, mask, aW1, ab1, aW2, ab2,
                                                hW1, hb1, hW2, hb2, (float*)d_out);
}
